// DTW_Model_91285234909681
// MI455X (gfx1250) — matrix operand order repaired and checked by reading, not compiled, not run on hardware
//
#include <hip/hip_runtime.h>
#include <hip/hip_bf16.h>

// ---------------------------------------------------------------------------
// DTW with cosine cost, MI455X (gfx1250, wave32).
//   B = 64, T1 = T2 = 512, D = 256, band W = 50, BIG = 1e9, EPS = 1e-8
//
// Stage 1: f32 -> f16 conversion + row L2 norms (wave reduction).
// Stage 2: band-restricted cost tiles via v_wmma_f32_16x16x32_f16.
//          For i-tile ti, only j-tiles ti-4 .. ti+4 intersect |i-j|<=50.
//          cost_ws[b][ti][u][16][16], u = jt - ti + 4 in [0,8].
// Stage 3: anti-diagonal DP, one wave32 per batch, diagonals in LDS,
//          no barriers (single-wave in-order LDS).
//
// Workspace layout (bytes):
//   x1h : 64*512*256*2 = 16 MiB
//   x2h : 64*512*256*2 = 16 MiB
//   n1  : 64*512*4
//   n2  : 64*512*4
//   cost: 64*32*9*256*4 = 18 MiB          total ~52.7 MB
// ---------------------------------------------------------------------------

#define BIGV 1e9f

typedef __attribute__((ext_vector_type(16))) _Float16 v16h;
typedef __attribute__((ext_vector_type(8)))  _Float16 h8;
typedef __attribute__((ext_vector_type(8)))  float    v8f;

union V16 { v16h v; h8 h[2]; };

// ---------------- Stage 1: convert to f16 + row norms ----------------------
// One wave (32 lanes) per row of 256 elements; 8 elements per lane.
__global__ __launch_bounds__(256) void cvt_norm_kernel(
    const float* __restrict__ x, _Float16* __restrict__ xh,
    float* __restrict__ nrm, int nrows) {
  int gtid = blockIdx.x * 256 + threadIdx.x;
  int row  = gtid >> 5;
  int lane = gtid & 31;
  if (row >= nrows) return;

  const float* r = x + (size_t)row * 256 + lane * 8;
  float4 a = ((const float4*)r)[0];
  float4 b = ((const float4*)r)[1];

  float ss = a.x * a.x;
  ss = fmaf(a.y, a.y, ss); ss = fmaf(a.z, a.z, ss); ss = fmaf(a.w, a.w, ss);
  ss = fmaf(b.x, b.x, ss); ss = fmaf(b.y, b.y, ss);
  ss = fmaf(b.z, b.z, ss); ss = fmaf(b.w, b.w, ss);

  h8 hv;
  hv[0] = (_Float16)a.x; hv[1] = (_Float16)a.y;
  hv[2] = (_Float16)a.z; hv[3] = (_Float16)a.w;
  hv[4] = (_Float16)b.x; hv[5] = (_Float16)b.y;
  hv[6] = (_Float16)b.z; hv[7] = (_Float16)b.w;
  *(h8*)(xh + (size_t)row * 256 + lane * 8) = hv;

  // wave32 reduction
  #pragma unroll
  for (int s = 16; s > 0; s >>= 1) ss += __shfl_xor(ss, s, 32);
  if (lane == 0) nrm[row] = fmaxf(sqrtf(ss), 1e-8f);
}

// ---------------- Stage 2: WMMA band cost tiles ----------------------------
// grid = (9, 32, 64); one wave per 16x16 tile; K=256 as 8 steps of 32.
// A-fragment (16x32 f16, ISA 7.12.2): lane m = lane&15, half = lane>>4,
//   v[0..7]  <- x1[row m][k0 + 8*half + 0..7]
//   v[8..15] <- x1[row m][k0 + 16 + 8*half + 0..7]
// B-fragment (32x16 f16): lane n = lane&15,
//   v[0..7]  <- x2[row n][k0 + 8*half + 0..7], v[8..15] <- x2[row n][k0 + 16 + 8*half + 0..7]   (B[k,n] = x2[n,k])
// D (16x16 f32): element e of lane holds (m = e + 8*half, n = lane&15).
__global__ __launch_bounds__(32) void cost_wmma_kernel(
    const _Float16* __restrict__ x1h, const _Float16* __restrict__ x2h,
    const float* __restrict__ n1, const float* __restrict__ n2,
    float* __restrict__ cost) {
  const int u  = blockIdx.x;            // 0..8
  const int ti = blockIdx.y;            // 0..31
  const int b  = blockIdx.z;            // 0..63
  const int tj = ti - 4 + u;
  if (tj < 0 || tj > 31) return;

  const int lane = threadIdx.x;
  const int half = lane >> 4;
  const int lo16 = lane & 15;

  const _Float16* arow = x1h + ((size_t)b * 512 + ti * 16 + lo16) * 256;
  const _Float16* brow = x2h + ((size_t)b * 512 + tj * 16 + lo16) * 256;

  v8f acc = {};
  #pragma unroll
  for (int k0 = 0; k0 < 256; k0 += 32) {
    V16 av, bv;
    av.h[0] = *(const h8*)(arow + k0 + 8 * half);
    av.h[1] = *(const h8*)(arow + k0 + 16 + 8 * half);
    bv.h[0] = *(const h8*)(brow + k0 + 8 * half);
    bv.h[1] = *(const h8*)(brow + k0 + 16 + 8 * half);
    acc = __builtin_amdgcn_wmma_f32_16x16x32_f16(
        false, av.v, false, bv.v, (short)0, acc, false, false);
  }

  const float rn2 = n2[b * 512 + tj * 16 + lo16];
  const float* n1p = n1 + b * 512 + ti * 16 + 8 * half;
  float* cout = cost + ((((size_t)b * 32 + ti) * 9 + u) << 8);

  #pragma unroll
  for (int e = 0; e < 8; ++e) {
    int m = e + 8 * half;
    float c = 1.0f - acc[e] / (n1p[e] * rn2);
    cout[(m << 4) + lo16] = c;
  }
}

// ---------------- Stage 3: banded DTW wavefront DP -------------------------
// One wave32 per batch; 2 cells per lane; LDS double buffer; no barriers
// (single wave => LDS ops are in program order).
__global__ __launch_bounds__(32) void dtw_kernel(
    const float* __restrict__ cost, float* __restrict__ out) {
  __shared__ float buf[2][512];
  const int b = blockIdx.x;
  const int lane = threadIdx.x;

  for (int i = lane; i < 512; i += 32) { buf[0][i] = BIGV; buf[1][i] = BIGV; }

  const float* cb = cost + (size_t)b * 32 * 9 * 256;

  for (int d = 0; d < 1022; ++d) {
    // active rows on diagonal d: i in [lo, hi], window of 64 slots covers it
    int lob = (d - 49) >> 1;                     // ceil((d-50)/2)
    int lo  = max(0, max(d - 511, lob));
    int ibase = max(0, lo - 2);

    float* p1 = buf[(d ^ 1) & 1];                // diagonal d-1
    float* pw = buf[d & 1];                      // holds d-2, becomes d

    float nv[2]; int ii[2];
    #pragma unroll
    for (int s = 0; s < 2; ++s) {
      int i = ibase + lane + s * 32;
      ii[s] = i;
      int j = d - i;
      float newv = BIGV;
      if (i < 512 && j >= 0 && j < 512 && (unsigned)(i - j + 50) <= 100u) {
        int tix = i >> 4;
        int ux  = (j >> 4) - tix + 4;            // in [0,8] for in-band cells
        float c = cb[((tix * 9 + ux) << 8) + ((i & 15) << 4) + (j & 15)];
        float m0 = p1[i];                        // acc[i,   j-1]
        float m1 = (i > 0) ? p1[i - 1] : BIGV;   // acc[i-1, j  ]
        float m2 = (i > 0) ? pw[i - 1] : BIGV;   // acc[i-1, j-1]
        float m = fminf(m0, fminf(m1, m2));
        if (d == 0) m = 0.0f;                    // DP origin (0,0)
        newv = c + m;
      }
      nv[s] = newv;
    }
    // all reads above are issued before these writes; per-wave LDS order
    #pragma unroll
    for (int s = 0; s < 2; ++s)
      if (ii[s] < 512) pw[ii[s]] = nv[s];        // inactive slots -> BIG

    if (d == 1021 && ii[0] == 511) out[b] = nv[0];   // acc[511,510]
  }
}

// ---------------------------------------------------------------------------
extern "C" void kernel_launch(void* const* d_in, const int* in_sizes, int n_in,
                              void* d_out, int out_size, void* d_ws, size_t ws_size,
                              hipStream_t stream) {
  (void)in_sizes; (void)n_in; (void)out_size; (void)ws_size;
  const float* x1 = (const float*)d_in[0];
  const float* x2 = (const float*)d_in[1];
  float* out = (float*)d_out;

  const size_t NEL = (size_t)64 * 512 * 256;     // 8,388,608 per tensor
  _Float16* x1h = (_Float16*)d_ws;
  _Float16* x2h = x1h + NEL;
  float* n1 = (float*)(x2h + NEL);
  float* n2 = n1 + 64 * 512;
  float* cost = n2 + 64 * 512;                   // 64*32*9*256 floats

  // Stage 1: convert + norms (32768 rows per tensor, 8 waves / block)
  const int rows = 64 * 512;
  const int cvtBlocks = (rows * 32) / 256;       // 4096
  cvt_norm_kernel<<<cvtBlocks, 256, 0, stream>>>(x1, x1h, n1, rows);
  cvt_norm_kernel<<<cvtBlocks, 256, 0, stream>>>(x2, x2h, n2, rows);

  // Stage 2: band cost tiles via WMMA
  cost_wmma_kernel<<<dim3(9, 32, 64), 32, 0, stream>>>(x1h, x2h, n1, n2, cost);

  // Stage 3: DTW DP, one wave per batch
  dtw_kernel<<<64, 32, 0, stream>>>(cost, out);
}
